// DoubleLayeredEncoder_53781580480949
// MI455X (gfx1250) — matrix operand order repaired and checked by reading, not compiled, not run on hardware
//
#include <hip/hip_runtime.h>
#include <hip/hip_bf16.h>

typedef __attribute__((ext_vector_type(16))) __bf16 v16bf;
typedef __attribute__((ext_vector_type(8)))  float  v8f;

#define GN     100000
#define GE     1600000
#define IN_CH  128
#define C1     128
#define HID    64

// ---------------------------------------------------------------------------
// Degree pipeline
// ---------------------------------------------------------------------------
__global__ void deg_init_kernel(float* __restrict__ deg1, float* __restrict__ deg2, int n) {
    int i = blockIdx.x * blockDim.x + threadIdx.x;
    if (i < n) { deg1[i] = 1.0f; deg2[i] = 1.0f; }   // self-loop weight 1 for both layers
}

__global__ void deg_accum_kernel(const int* __restrict__ ei, const float* __restrict__ w,
                                 const int* __restrict__ et,
                                 float* __restrict__ deg1, float* __restrict__ deg2,
                                 float* __restrict__ wet, int ne) {
    int e = blockIdx.x * blockDim.x + threadIdx.x;
    if (e < ne) {
        int dst = ei[ne + e];
        float etf = (float)et[e];
        wet[e] = etf;                       // cache edge_type as float for layer-2 scatter
        atomicAdd(deg1 + dst, w[e]);
        atomicAdd(deg2 + dst, etf);
    }
}

__global__ void deg_rsqrt_kernel(float* __restrict__ deg1, float* __restrict__ deg2, int n) {
    int i = blockIdx.x * blockDim.x + threadIdx.x;
    if (i < n) {
        float d1 = deg1[i]; deg1[i] = (d1 > 0.0f) ? rsqrtf(d1) : 0.0f;
        float d2 = deg2[i]; deg2[i] = (d2 > 0.0f) ? rsqrtf(d2) : 0.0f;
    }
}

// ---------------------------------------------------------------------------
// bf16 WMMA GEMM:  Y[M x OUT] = X[M x IN] @ W[IN x OUT]   (no bias here)
// one wave -> one 16x16 output tile; 8 waves/block -> 128 rows/block
// ---------------------------------------------------------------------------
template<int IN, int OUT>
__global__ void gemm_wmma_bf16(const float* __restrict__ X, const float* __restrict__ W,
                               float* __restrict__ Y, int M) {
    const int lane    = threadIdx.x & 31;
    const int wave    = threadIdx.x >> 5;
    const int mbase   = (blockIdx.x * 8 + wave) * 16;
    const int nbase   = blockIdx.y * 16;
    if (mbase >= M) return;                 // wave-uniform: EXEC stays all-1 for WMMA

    const int halfSel = lane >> 4;          // 0: lanes 0-15, 1: lanes 16-31
    const int l16     = lane & 15;
    int row = mbase + l16; if (row >= M) row = M - 1;   // clamp loads; stores guarded
    const int col = nbase + l16;

    const int kbA = halfSel * 16;           // A layout: contiguous k = 16*half + s (matches B)
    const int kbB = halfSel * 16;           // B layout: 16-bit B 32x16

    v8f acc = {};
#pragma unroll
    for (int k0 = 0; k0 < IN; k0 += 32) {
        v16bf a, b;
        const float* xr = X + (size_t)row * IN + k0 + kbA;
#pragma unroll
        for (int j = 0; j < 8; ++j) a[j]     = (__bf16)xr[j];       // K = k0+kb+0..7
#pragma unroll
        for (int j = 0; j < 8; ++j) a[8 + j] = (__bf16)xr[8 + j];   // K = k0+kb+8..15
        const float* wc = W + (size_t)(k0 + kbB) * OUT + col;
#pragma unroll
        for (int j = 0; j < 16; ++j) b[j] = (__bf16)wc[(size_t)j * OUT];
        acc = __builtin_amdgcn_wmma_f32_16x16x32_bf16(false, a, false, b,
                                                      (short)0, acc, false, false);
    }

    // D layout: VGPR r -> M = r (lanes 0-15) / 8+r (lanes 16-31), N = lane&15
#pragma unroll
    for (int r = 0; r < 8; ++r) {
        int orow = mbase + r + 8 * halfSel;
        if (orow < M) Y[(size_t)orow * OUT + col] = acc[r];
    }
}

// ---------------------------------------------------------------------------
// Aggregation
// ---------------------------------------------------------------------------
template<int C>
__global__ void self_loop_init_kernel(const float* __restrict__ xw, const float* __restrict__ dinv,
                                      float* __restrict__ agg, int n) {
    int i = blockIdx.x * blockDim.x + threadIdx.x;    // over n*C (<= 12.8M)
    if (i < n * C) {
        float di = dinv[i / C];                       // norm(self) = dinv*1*dinv
        agg[i] = xw[i] * di * di;
    }
}

// one wave per edge; C/32 channels per lane, contiguous -> fully coalesced gather
template<int C>
__global__ void edge_agg_kernel(const int* __restrict__ ei, const float* __restrict__ wgt,
                                const float* __restrict__ dinv, const float* __restrict__ xw,
                                float* __restrict__ agg, int ne) {
    int t = blockIdx.x * blockDim.x + threadIdx.x;
    int e = t >> 5;
    int lane = t & 31;
    if (e >= ne) return;
    int src = ei[e];
    int dst = ei[ne + e];
    float norm = dinv[src] * wgt[e] * dinv[dst];
    constexpr int V = C / 32;
    const float* s = xw  + (size_t)src * C + lane * V;
    float*       d = agg + (size_t)dst * C + lane * V;
#pragma unroll
    for (int j = 0; j < V; ++j) atomicAdd(d + j, s[j] * norm);
}

template<int C>
__global__ void bias_prelu_kernel(const float* __restrict__ agg, const float* __restrict__ b,
                                  const float* __restrict__ a, float* __restrict__ out, int n) {
    int i = blockIdx.x * blockDim.x + threadIdx.x;
    if (i < n * C) {
        int c = i & (C - 1);
        float v = agg[i] + b[c];
        out[i] = (v >= 0.0f) ? v : a[c] * v;
    }
}

// fused: bias + PReLU on both halves + average
__global__ void final_combine_kernel(const float* __restrict__ agg2, const float* __restrict__ b2,
                                     const float* __restrict__ a2, float* __restrict__ out, int half) {
    int i = blockIdx.x * blockDim.x + threadIdx.x;    // over half*HID (3.2M)
    if (i < half * HID) {
        int c = i & (HID - 1);
        float bb = b2[c], aa = a2[c];
        float v1 = agg2[i] + bb;                 v1 = (v1 >= 0.0f) ? v1 : aa * v1;
        float v2 = agg2[i + half * HID] + bb;    v2 = (v2 >= 0.0f) ? v2 : aa * v2;
        out[i] = 0.5f * (v1 + v2);
    }
}

// ---------------------------------------------------------------------------
extern "C" void kernel_launch(void* const* d_in, const int* in_sizes, int n_in,
                              void* d_out, int out_size, void* d_ws, size_t ws_size,
                              hipStream_t stream) {
    const float* x   = (const float*)d_in[0];
    const int*   ei  = (const int*)  d_in[1];   // [2, E]
    const float* ew  = (const float*)d_in[2];
    const int*   et  = (const int*)  d_in[3];
    const float* W1  = (const float*)d_in[4];
    const float* b1  = (const float*)d_in[5];
    const float* a1  = (const float*)d_in[6];
    const float* W2  = (const float*)d_in[7];
    const float* b2  = (const float*)d_in[8];
    const float* a2  = (const float*)d_in[9];
    float* out = (float*)d_out;

    // workspace layout (floats)
    float* ws   = (float*)d_ws;
    float* deg1 = ws;                         // N   (becomes dinv1)
    float* deg2 = deg1 + GN;                  // N   (becomes dinv2)
    float* wet  = deg2 + GN;                  // E   (edge_type as float)
    float* bufA = wet  + GE;                  // N*128 : xw1, then h1
    float* bufB = bufA + (size_t)GN * C1;     // N*128 : agg1; later xw2 | agg2
    float* xw2  = bufB;                       // N*64
    float* agg2 = bufB + (size_t)GN * HID;    // N*64

    const int TB = 256;
    // --- degrees ---
    deg_init_kernel <<<(GN + TB - 1) / TB, TB, 0, stream>>>(deg1, deg2, GN);
    deg_accum_kernel<<<(GE + TB - 1) / TB, TB, 0, stream>>>(ei, ew, et, deg1, deg2, wet, GE);
    deg_rsqrt_kernel<<<(GN + TB - 1) / TB, TB, 0, stream>>>(deg1, deg2, GN);

    // --- layer 1 ---
    dim3 g1((GN + 127) / 128, C1 / 16);
    gemm_wmma_bf16<IN_CH, C1><<<g1, TB, 0, stream>>>(x, W1, bufA, GN);
    self_loop_init_kernel<C1><<<(GN * C1 + TB - 1) / TB, TB, 0, stream>>>(bufA, deg1, bufB, GN);
    edge_agg_kernel<C1><<<(GE * 32) / TB, TB, 0, stream>>>(ei, ew, deg1, bufA, bufB, GE);
    bias_prelu_kernel<C1><<<(GN * C1 + TB - 1) / TB, TB, 0, stream>>>(bufB, b1, a1, bufA, GN);

    // --- layer 2 ---
    dim3 g2((GN + 127) / 128, HID / 16);
    gemm_wmma_bf16<C1, HID><<<g2, TB, 0, stream>>>(bufA, W2, xw2, GN);
    self_loop_init_kernel<HID><<<(GN * HID + TB - 1) / TB, TB, 0, stream>>>(xw2, deg2, agg2, GN);
    edge_agg_kernel<HID><<<(GE * 32) / TB, TB, 0, stream>>>(ei, wet, deg2, xw2, agg2, GE);

    // --- output ---
    const int half = GN / 2;
    final_combine_kernel<<<(half * HID + TB - 1) / TB, TB, 0, stream>>>(agg2, b2, a2, out, half);
}
